// HeadLayer_84301618086492
// MI455X (gfx1250) — matrix operand order repaired and checked by reading, not compiled, not run on hardware
//
#include <hip/hip_runtime.h>

// ---------------------------------------------------------------------------
// Single-head causal attention for MI455X (gfx1250, wave32, WMMA).
//   x:[8,2048,1024]f32  Wq/Wk/Wv:[1024,64]f32  ->  out:[8,2048,64]f32
// Path: f32 -> bf16 WMMA 16x16x32 (f32 accum) flash attention with
// ASYNCcnt-tracked GLOBAL_LOAD_ASYNC_TO_LDS double-buffered K/V staging.
// ---------------------------------------------------------------------------

typedef __bf16 bf16_t;
typedef __attribute__((ext_vector_type(16))) __bf16 v16bf;
typedef __attribute__((ext_vector_type(8)))  __bf16 v8bf;
typedef __attribute__((ext_vector_type(8)))  float  v8f;

union V16 { v16bf v; v8bf h[2]; };

__device__ __forceinline__ bf16_t f2bf(float f) { return (bf16_t)f; }

__device__ __forceinline__ v8f wmma_bf16(v16bf a, v16bf b, v8f c) {
  // D = A(16x32 bf16) * B(32x16 bf16) + C(16x16 f32)
  return __builtin_amdgcn_wmma_f32_16x16x32_bf16(
      /*neg_a=*/false, a, /*neg_b=*/false, b,
      /*c_mod=*/(short)0, c, /*reuse_a=*/false, /*reuse_b=*/false);
}

// Async memory -> LDS copy, 16 bytes per lane. LDS byte offset is the low
// 32 bits of the generic pointer.
__device__ __forceinline__ void async_cp16(bf16_t* lds_dst, const bf16_t* gsrc) {
  unsigned loff = (unsigned)(unsigned long long)(uintptr_t)lds_dst;
  asm volatile("global_load_async_to_lds_b128 %0, %1, off"
               :: "v"(loff), "v"(gsrc) : "memory");
}

__device__ __forceinline__ void wait_async0() {
#if __has_builtin(__builtin_amdgcn_s_wait_asynccnt)
  __builtin_amdgcn_s_wait_asynccnt(0);
#else
  asm volatile("s_wait_asynccnt 0" ::: "memory");
#endif
}

#define EMB   1024
#define HSZ   64
#define SEQ   2048
#define BATCH 8
#define QTILE 128
#define KTILE 64

// ---------------------------------------------------------------------------
// Kernel 0: W[e][h] f32  ->  WT[m][h][e] bf16   (B-fragment friendly)
// ---------------------------------------------------------------------------
__global__ void __launch_bounds__(256)
wt_kernel(const float* __restrict__ Wq, const float* __restrict__ Wk,
          const float* __restrict__ Wv, bf16_t* __restrict__ WT) {
  int idx = blockIdx.x * 256 + threadIdx.x;
  if (idx >= 3 * EMB * HSZ) return;
  int m   = idx / (EMB * HSZ);
  int rem = idx % (EMB * HSZ);
  int e = rem / HSZ, h = rem % HSZ;
  const float* W = (m == 0) ? Wq : ((m == 1) ? Wk : Wv);
  WT[((size_t)m * HSZ + h) * EMB + e] = f2bf(W[(size_t)e * HSZ + h]);
}

// ---------------------------------------------------------------------------
// Kernel 1: projections.  Block = 128 rows (8 waves x 16 rows).
//   Qb,Kb : [B*S][64] bf16 row-major
//   Vt    : [B][64][S] bf16 (transposed for P*V B-fragments)
// ---------------------------------------------------------------------------
__global__ void __launch_bounds__(256)
proj_kernel(const float* __restrict__ x, const bf16_t* __restrict__ WT,
            bf16_t* __restrict__ Qb, bf16_t* __restrict__ Kb,
            bf16_t* __restrict__ Vt) {
  const int wave = threadIdx.x >> 5;
  const int lane = threadIdx.x & 31;
  const int l16  = lane & 15;
  const int half = lane >> 4;
  const int b0   = half * 16;
  const int row0 = blockIdx.x * 128 + wave * 16;   // wave's 16 rows
  const int arow = row0 + l16;                     // A-fragment row (M = l16)

  v8f acc[3][4];
#pragma unroll
  for (int m = 0; m < 3; ++m)
#pragma unroll
    for (int n = 0; n < 4; ++n) acc[m][n] = (v8f)0.0f;

  for (int kc = 0; kc < EMB; kc += 32) {
    // A fragment: x rows (f32 -> bf16). elems 0..15 = K b0..b0+15
    const float* xr = x + (size_t)arow * EMB + kc + b0;
    v8f f0 = *(const v8f*)(xr);
    v8f f1 = *(const v8f*)(xr + 8);
    V16 a;
#pragma unroll
    for (int j = 0; j < 8; ++j) { a.h[0][j] = f2bf(f0[j]); a.h[1][j] = f2bf(f1[j]); }

#pragma unroll
    for (int m = 0; m < 3; ++m) {
#pragma unroll
      for (int n = 0; n < 4; ++n) {
        const int hcol = n * 16 + l16;             // B lane = output column
        V16 b;
        b.v = *(const v16bf*)(WT + ((size_t)m * HSZ + hcol) * EMB + kc + half * 16);
        acc[m][n] = wmma_bf16(a.v, b.v, acc[m][n]);
      }
    }
  }

  // Store (C layout: VGPR r -> row = row0 + r + 8*half, col = n*16 + l16)
#pragma unroll
  for (int n = 0; n < 4; ++n) {
#pragma unroll
    for (int r = 0; r < 8; ++r) {
      const int srow = row0 + r + 8 * half;
      const int hcol = n * 16 + l16;
      Qb[(size_t)srow * HSZ + hcol] = f2bf(acc[0][n][r]);
      Kb[(size_t)srow * HSZ + hcol] = f2bf(acc[1][n][r]);
      const int bb = srow / SEQ, s = srow % SEQ;
      Vt[((size_t)bb * HSZ + hcol) * SEQ + s] = f2bf(acc[2][n][r]);
    }
  }
}

// ---------------------------------------------------------------------------
// Kernel 2: flash attention.  Block = 1 batch x 128 queries (8 waves x 16).
// Double-buffered async K/V staging + online softmax in C-layout halves.
// ---------------------------------------------------------------------------
__global__ void __launch_bounds__(256)
attn_kernel(const bf16_t* __restrict__ Qb, const bf16_t* __restrict__ Kb,
            const bf16_t* __restrict__ Vt, float* __restrict__ out) {
  __shared__ bf16_t lK[2][KTILE * HSZ];     // [key][h]   2 x 8 KB
  __shared__ bf16_t lV[2][HSZ * KTILE];     // [h][key]   2 x 8 KB
  __shared__ bf16_t lP[8 * 16 * KTILE];     // per-wave      16 KB

  const int wave = threadIdx.x >> 5;
  const int lane = threadIdx.x & 31;
  const int l16  = lane & 15;
  const int half = lane >> 4;
  const int b0   = half * 16;

  const int bq = blockIdx.x / (SEQ / QTILE);
  const int q0 = (blockIdx.x % (SEQ / QTILE)) * QTILE;
  const int qw = q0 + wave * 16;

  const bf16_t* Qbase = Qb + (size_t)bq * SEQ * HSZ;
  const bf16_t* Kbase = Kb + (size_t)bq * SEQ * HSZ;
  const bf16_t* Vbase = Vt + (size_t)bq * HSZ * SEQ;

  // Q A-fragments (held in registers for the whole kernel), 2 chunks over H=64.
  V16 qa[2];
  const int arow = qw + l16;
#pragma unroll
  for (int c = 0; c < 2; ++c) {
    const bf16_t* qp = Qbase + (size_t)arow * HSZ + c * 32 + b0;
    qa[c].h[0] = *(const v8bf*)(qp);
    qa[c].h[1] = *(const v8bf*)(qp + 8);
  }

  v8f O[4];
#pragma unroll
  for (int hn = 0; hn < 4; ++hn) O[hn] = (v8f)0.0f;
  float mrow[8], lrow[8];
#pragma unroll
  for (int r = 0; r < 8; ++r) { mrow[r] = -__builtin_inff(); lrow[r] = 0.0f; }

  const float scale = 0.03125f;                    // EMB^-0.5 = 1/32
  const int ntiles = (q0 + QTILE) / KTILE;         // causal bound
  const int per_row = KTILE / 8;                   // 16B units per V^T row

  // --- prologue: async-stage tile 0 into buffer 0 ---
#pragma unroll
  for (int i = threadIdx.x; i < KTILE * HSZ / 8; i += 256)
    async_cp16(&lK[0][i * 8], Kbase + i * 8);
#pragma unroll
  for (int i = threadIdx.x; i < HSZ * per_row; i += 256) {
    const int h = i / per_row, j = i % per_row;
    async_cp16(&lV[0][h * KTILE + j * 8], Vbase + (size_t)h * SEQ + j * 8);
  }

  int cur = 0;
  for (int kt = 0; kt < ntiles; ++kt) {
    const int k0 = kt * KTILE;
    wait_async0();                 // my in-flight loads for tile kt landed
    __syncthreads();               // everyone's landed; prev compute finished

    // issue async loads for tile kt+1 into the other buffer
    if (kt + 1 < ntiles) {
      const int kn = k0 + KTILE;
      const int nxt = cur ^ 1;
#pragma unroll
      for (int i = threadIdx.x; i < KTILE * HSZ / 8; i += 256)
        async_cp16(&lK[nxt][i * 8], Kbase + (size_t)kn * HSZ + i * 8);
#pragma unroll
      for (int i = threadIdx.x; i < HSZ * per_row; i += 256) {
        const int h = i / per_row, j = i % per_row;
        async_cp16(&lV[nxt][h * KTILE + j * 8],
                   Vbase + (size_t)h * SEQ + kn + j * 8);
      }
    }

    const bf16_t* cK = lK[cur];
    const bf16_t* cV = lV[cur];

    // ---- S = Q * K^T  (4 N-tiles of 16 keys, K-dim = H in 2 chunks) ----
    v8f sa[4];
#pragma unroll
    for (int n = 0; n < 4; ++n) {
      sa[n] = (v8f)0.0f;
      const int key = n * 16 + l16;                // B lane = key column
#pragma unroll
      for (int c = 0; c < 2; ++c) {
        V16 kb;
        kb.v = *(const v16bf*)(cK + key * HSZ + c * 32 + half * 16);
        sa[n] = wmma_bf16(qa[c].v, kb.v, sa[n]);
      }
    }

    // ---- scale + causal mask + row max ----
    float rmax[8];
#pragma unroll
    for (int r = 0; r < 8; ++r) rmax[r] = -__builtin_inff();
#pragma unroll
    for (int n = 0; n < 4; ++n) {
      const int ki = k0 + n * 16 + l16;
#pragma unroll
      for (int r = 0; r < 8; ++r) {
        const int qi = qw + r + 8 * half;
        float s = sa[n][r] * scale;
        if (ki > qi) s = -__builtin_inff();
        sa[n][r] = s;
        rmax[r] = fmaxf(rmax[r], s);
      }
    }
#pragma unroll
    for (int msk = 1; msk <= 8; msk <<= 1)
#pragma unroll
      for (int r = 0; r < 8; ++r)
        rmax[r] = fmaxf(rmax[r], __shfl_xor(rmax[r], msk, 32));

    // ---- online softmax update ----
    float corr[8];
#pragma unroll
    for (int r = 0; r < 8; ++r) {
      const float mn = fmaxf(mrow[r], rmax[r]);
      corr[r] = __expf(mrow[r] - mn);              // exp(-inf)=0 on first tile
      mrow[r] = mn;
    }
    float rsum[8];
#pragma unroll
    for (int r = 0; r < 8; ++r) rsum[r] = 0.0f;
#pragma unroll
    for (int n = 0; n < 4; ++n)
#pragma unroll
      for (int r = 0; r < 8; ++r) {
        const float p = __expf(sa[n][r] - mrow[r]);
        sa[n][r] = p;
        rsum[r] += p;
      }
#pragma unroll
    for (int msk = 1; msk <= 8; msk <<= 1)
#pragma unroll
      for (int r = 0; r < 8; ++r) rsum[r] += __shfl_xor(rsum[r], msk, 32);
#pragma unroll
    for (int r = 0; r < 8; ++r) lrow[r] = lrow[r] * corr[r] + rsum[r];
#pragma unroll
    for (int hn = 0; hn < 4; ++hn)
#pragma unroll
      for (int r = 0; r < 8; ++r) O[hn][r] *= corr[r];

    // ---- P -> LDS (bf16, per-wave region, row-major [16][KTILE]) ----
    bf16_t* Pw = lP + wave * 16 * KTILE;
#pragma unroll
    for (int n = 0; n < 4; ++n)
#pragma unroll
      for (int r = 0; r < 8; ++r)
        Pw[(r + 8 * half) * KTILE + n * 16 + l16] = f2bf(sa[n][r]);

    // ---- O += P * V  (K-dim = 64 keys in 2 chunks of 32) ----
#pragma unroll
    for (int c = 0; c < 2; ++c) {
      V16 pa;
      const bf16_t* pp = Pw + l16 * KTILE + c * 32 + b0;
      pa.h[0] = *(const v8bf*)(pp);
      pa.h[1] = *(const v8bf*)(pp + 8);
#pragma unroll
      for (int hn = 0; hn < 4; ++hn) {
        V16 vb;
        vb.v = *(const v16bf*)(cV + (hn * 16 + l16) * KTILE + c * 32 + half * 16);
        O[hn] = wmma_bf16(pa.v, vb.v, O[hn]);
      }
    }

    cur ^= 1;
  }

  // ---- epilogue: out = O / l ----
  float* orow = out + (size_t)bq * SEQ * HSZ;
#pragma unroll
  for (int hn = 0; hn < 4; ++hn)
#pragma unroll
    for (int r = 0; r < 8; ++r) {
      const int qi = qw + r + 8 * half;
      orow[(size_t)qi * HSZ + hn * 16 + l16] = O[hn][r] / lrow[r];
    }
}

// ---------------------------------------------------------------------------
extern "C" void kernel_launch(void* const* d_in, const int* in_sizes, int n_in,
                              void* d_out, int out_size, void* d_ws, size_t ws_size,
                              hipStream_t stream) {
  (void)in_sizes; (void)n_in; (void)out_size; (void)ws_size;
  const float* x  = (const float*)d_in[0];
  const float* Wq = (const float*)d_in[1];
  const float* Wk = (const float*)d_in[2];
  const float* Wv = (const float*)d_in[3];

  // Workspace layout (bf16):
  //   WT : 3*64*1024            = 384 KB
  //   Qb : 8*2048*64            =   2 MB
  //   Kb : 8*2048*64            =   2 MB
  //   Vt : 8*64*2048            =   2 MB   (total ~6.4 MB)
  char* ws = (char*)d_ws;
  bf16_t* WT = (bf16_t*)(ws);
  bf16_t* Qb = (bf16_t*)(ws + 393216);
  bf16_t* Kb = (bf16_t*)(ws + 393216 + 2097152);
  bf16_t* Vt = (bf16_t*)(ws + 393216 + 2 * 2097152);

  wt_kernel<<<768, 256, 0, stream>>>(Wq, Wk, Wv, WT);
  proj_kernel<<<BATCH * SEQ / 128, 256, 0, stream>>>(x, WT, Qb, Kb, Vt);
  attn_kernel<<<BATCH * (SEQ / QTILE), 256, 0, stream>>>(Qb, Kb, Vt, (float*)d_out);
}
